// PartitionedAutoDecoder_76828374991372
// MI455X (gfx1250) — compile-verified, matrix operand order checked by reading, not run on hardware
//
#include <hip/hip_runtime.h>
#include <hip/hip_bf16.h>

typedef __attribute__((ext_vector_type(16))) _Float16 v16h;
typedef __attribute__((ext_vector_type(8)))  float    v8f;
typedef __attribute__((ext_vector_type(4)))  unsigned int u32x4;
typedef __attribute__((ext_vector_type(8)))  int        i32x8;
typedef __attribute__((ext_vector_type(4)))  int        i32x4;
typedef _Float16 half_t;

#define PS 8
#define TILE_ROWS 128
#define WAVES 8

// Weight fragment regions in d_ws (units: halfs). Each fragment chunk = 512 halfs
// (one 32x16 or 16x32 WMMA operand striped across 32 lanes x 16 halfs).
#define W_INIT_OFF 0        // 3 chunks x 4 tiles x 512 = 6144
#define W0A_OFF    6144     // 2 x 4 x 512 = 4096
#define W0B_OFF    10240    // 3 x 4 x 512 = 6144
#define W1_OFF     16384    // 4096
#define W2_OFF     20480    // 4096
#define W3_OFF     24576    // 4096
#define WOUT_OFF   28672    // 2 x 1 x 512 = 1024
#define WTOT       29696    // halfs (59392 bytes)

// A-matrix (16x32 f16) lane/half -> K mapping (ISA 7.12.2):
//   lane group g = lane>>4, half index i: K = (i>>3)*16 + g*8 + (i&7), M = lane&15
// frag_off gives the half-offset within a 512-half chunk for element (m, k):
__device__ __forceinline__ int frag_off(int m, int k) {
  int g = (k & 15) >> 3;
  return ((g << 4) + m) * 16 + ((k >> 4) << 3) + (k & 7);
}

// ---------------- prep kernel: f32 weights -> f16 fragment-ordered in d_ws ----
__global__ __launch_bounds__(256)
void prep_weights_kernel(const float* __restrict__ Wi,  const float* __restrict__ W0,
                         const float* __restrict__ W1,  const float* __restrict__ W2,
                         const float* __restrict__ W3,  const float* __restrict__ Wo,
                         half_t* __restrict__ ws) {
  int i = blockIdx.x * blockDim.x + threadIdx.x;
  if (i >= WTOT) return;
  const float* W; int base, tiles, rows, rowoff, ldw, cols;
  if (i < W0A_OFF)       { W = Wi; base = W_INIT_OFF; tiles = 4; rows = 83;  rowoff = 0;  ldw = 64; cols = 64; }
  else if (i < W0B_OFF)  { W = W0; base = W0A_OFF;    tiles = 4; rows = 64;  rowoff = 0;  ldw = 64; cols = 64; }
  else if (i < W1_OFF)   { W = W0; base = W0B_OFF;    tiles = 4; rows = 147; rowoff = 64; ldw = 64; cols = 64; }
  else if (i < W2_OFF)   { W = W1; base = W1_OFF;     tiles = 4; rows = 64;  rowoff = 0;  ldw = 64; cols = 64; }
  else if (i < W3_OFF)   { W = W2; base = W2_OFF;     tiles = 4; rows = 64;  rowoff = 0;  ldw = 64; cols = 64; }
  else if (i < WOUT_OFF) { W = W3; base = W3_OFF;     tiles = 4; rows = 64;  rowoff = 0;  ldw = 64; cols = 64; }
  else                   { W = Wo; base = WOUT_OFF;   tiles = 1; rows = 64;  rowoff = 0;  ldw = 3;  cols = 3;  }
  int e      = i - base;
  int frag   = e >> 9;
  int within = e & 511;
  int lane   = within >> 4;
  int h      = within & 15;
  int chunk  = frag / tiles;
  int nt     = frag % tiles;
  int g      = lane >> 4;
  int n0     = lane & 15;
  // B-matrix (32x16 f16) assumed lane/half mapping (mirror of A, ISA 7.12.2):
  //   N = lane&15, K = (h>>3)*16 + g*8 + (h&7)
  int kval = (h >> 3) * 16 + g * 8 + (h & 7);
  int k    = rowoff + chunk * 32 + kval;
  int col  = nt * 16 + n0;
  float v  = (k < rows && col < cols) ? W[k * ldw + col] : 0.0f;
  ws[i] = (half_t)v;
}

// ---------------- main kernel helpers ----------------------------------------
template <int NCH, int NTILES, bool ACT>
__device__ __forceinline__ void mm_frag(const half_t* __restrict__ abuf,
                                        const half_t* __restrict__ wbase,
                                        int lane, v8f* acc) {
#pragma unroll
  for (int c = 0; c < NCH; ++c) {
    v16h a = *(const v16h*)(abuf + c * 512 + lane * 16);
    if (ACT) {
#pragma unroll
      for (int i = 0; i < 16; ++i) {
        half_t h = a[i];
        half_t s = h * (half_t)0.01f;
        a[i] = (h > s) ? h : s;   // leaky_relu = max(x, 0.01x)
      }
    }
#pragma unroll
    for (int t = 0; t < NTILES; ++t) {
      v16h b = *(const v16h*)(wbase + (c * NTILES + t) * 512 + lane * 16);
      acc[t] = __builtin_amdgcn_wmma_f32_16x16x32_f16(
          false, a, false, b, (short)0, acc[t], false, false);
    }
  }
}

// bias + leaky_relu + store into fragment-ordered f16 activation buffer
__device__ __forceinline__ void store_act64(v8f* acc, const float* __restrict__ bias,
                                            half_t* __restrict__ xbuf, int lane) {
  int n0 = lane & 15, g = lane >> 4;
#pragma unroll
  for (int t = 0; t < 4; ++t) {
    int n = t * 16 + n0;
    float bv = bias[n];
#pragma unroll
    for (int r = 0; r < 8; ++r) {
      int m = r + g * 8;
      float v = acc[t][r] + bv;
      v = fmaxf(v, 0.01f * v);
      xbuf[(n >> 5) * 512 + frag_off(m, n & 31)] = (half_t)v;
    }
  }
}

// ---------------- main kernel -------------------------------------------------
__global__ __launch_bounds__(256)
void pad_mlp_kernel(const float* __restrict__ p, const float* __restrict__ code,
                    const float* __restrict__ basis,
                    const float* __restrict__ b_init, const float* __restrict__ b0,
                    const float* __restrict__ b1, const float* __restrict__ b2,
                    const float* __restrict__ b3, const float* __restrict__ b_out,
                    const half_t* __restrict__ wprep, float* __restrict__ out, int N) {
  __shared__ half_t sW[WTOT];                 // 59392 B, all weights, frag order
  __shared__ half_t sFeat[WAVES][3 * 512];    // 24576 B, init features (raw f16)
  __shared__ half_t sX[WAVES][2 * 512];       // 16384 B, layer activations
  int tid  = threadIdx.x;
  int lane = tid & 31;
  int wave = tid >> 5;

#if __has_builtin(__builtin_amdgcn_tensor_load_to_lds)
  // ---- async weight stage: one TDM descriptor copies the 59392B fragment-
  // ordered weight blob (1 row of 7424 x 8B elements) from d_ws into LDS.
  // D# layout per CDNA5 ISA ch.8: group0 = {count/flags, lds_addr,
  // global_addr[56:0], type=2}; group1 = {mask/data_size, dims/tile dims,
  // strides}; groups 2/3 zero (dims beyond 2D unused).
  if (wave == 0) {
    unsigned long long ga = (unsigned long long)(uintptr_t)wprep;
    unsigned int lds_base = (unsigned int)(uintptr_t)sW;  // low 32 bits = LDS offset
    const unsigned int nelem = WTOT * 2 / 8;              // 7424 8-byte elements
    u32x4 g0;
    g0[0] = 1u;                                   // count=1 (valid descriptor)
    g0[1] = lds_base;                             // lds_addr  (bits 63:32)
    g0[2] = (unsigned int)(ga & 0xffffffffu);     // global_addr low
    g0[3] = (unsigned int)(ga >> 32) | (2u << 30);// global_addr high | type=2
    i32x8 g1;
    g1[0] = (int)(3u << 16);                      // wg_mask=0, data_size=3 (8B)
    g1[1] = (int)((nelem & 0xffffu) << 16);       // tensor_dim0 low16 @bit48
    g1[2] = (int)(((nelem >> 16) & 0xffffu) | (1u << 16)); // dim0 hi | tensor_dim1=1
    g1[3] = (int)(nelem << 16);                   // tensor_dim1 hi=0 | tile_dim0
    g1[4] = 1;                                    // tile_dim1=1, tile_dim2=0
    g1[5] = (int)nelem;                           // tensor_dim0_stride low32
    g1[6] = 0;                                    // stride hi | dim1_stride lo
    g1[7] = 0;                                    // dim1_stride hi
    i32x4 gz4 = (i32x4){0, 0, 0, 0};
    i32x8 gz8 = (i32x8){0, 0, 0, 0, 0, 0, 0, 0};
    __builtin_amdgcn_tensor_load_to_lds(g0, g1, gz4, gz4, gz8, 0);
  }
#else
  {
    const uint4* src = (const uint4*)wprep;
    uint4* dst = (uint4*)sW;
#pragma unroll 4
    for (int j = tid; j < WTOT / 8; j += 256) dst[j] = src[j];
  }
#endif

  int row0 = blockIdx.x * TILE_ROWS + wave * 16;
  half_t* feat = sFeat[wave];
  half_t* xb   = sX[wave];

  // ---- per-row point features (overlapped with the weight DMA) ----
  // lanes 0..15 each own row m = lane
  int m = lane & 15;
  int row = row0 + m;
  float l0 = 0.f, l1 = 0.f, l2 = 0.f;
  int idx = 0;
  if (lane < 16 && row < N) {
    float p0 = p[row * 3 + 0], p1 = p[row * 3 + 1], p2 = p[row * 3 + 2];
    float c0 = fminf(fmaxf(floorf(p0 * (float)PS), 0.f), (float)(PS - 1));
    float c1 = fminf(fmaxf(floorf(p1 * (float)PS), 0.f), (float)(PS - 1));
    float c2 = fminf(fmaxf(floorf(p2 * (float)PS), 0.f), (float)(PS - 1));
    idx = (int)c0 + PS * (int)c1 + PS * PS * (int)c2;
    l0 = p0 - (c0 + 0.5f) * (1.0f / PS);
    l1 = p1 - (c1 + 0.5f) * (1.0f / PS);
    l2 = p2 - (c2 + 0.5f) * (1.0f / PS);
  }

  // code gather: all 32 lanes fetch code[idx_m][lane] and [32+lane] (coalesced)
  for (int mm = 0; mm < 16; ++mm) {
    int im = __shfl(idx, mm);
    const float* cr = code + im * 64;
    feat[frag_off(mm, lane)]       = (half_t)cr[lane];        // K chunk 0 (k=lane)
    feat[512 + frag_off(mm, lane)] = (half_t)cr[32 + lane];   // K chunk 1
  }

  // chunk 2: local(3) + cos(8) + sin(8) + zero pad (k=64..95 absolute)
  if (lane < 16) {
    half_t* f2 = feat + 2 * 512;
    f2[frag_off(m, 0)] = (half_t)l0;
    f2[frag_off(m, 1)] = (half_t)l1;
    f2[frag_off(m, 2)] = (half_t)l2;
#pragma unroll
    for (int j = 0; j < 8; ++j) {
      float xbv = l0 * basis[0 * 8 + j] + l1 * basis[1 * 8 + j] + l2 * basis[2 * 8 + j];
      f2[frag_off(m, 3 + j)]  = (half_t)__cosf(xbv);
      f2[frag_off(m, 11 + j)] = (half_t)__sinf(xbv);
    }
#pragma unroll
    for (int k = 19; k < 32; ++k) f2[frag_off(m, k)] = (half_t)0.f;
  }

#if __has_builtin(__builtin_amdgcn_tensor_load_to_lds)
  __builtin_amdgcn_s_wait_tensorcnt(0);  // wave0: DMA complete; others: no-op
#endif
  __syncthreads();                        // weights visible to all waves

  v8f acc[4];

  // x = init_feat @ W_init + b_init ; stored as act(x)
#pragma unroll
  for (int t = 0; t < 4; ++t) acc[t] = (v8f){0.f, 0.f, 0.f, 0.f, 0.f, 0.f, 0.f, 0.f};
  mm_frag<3, 4, false>(feat, sW + W_INIT_OFF, lane, acc);
  store_act64(acc, b_init, xb, lane);

  // layer 0: act([x, feat]) @ W0 + b0  (split: act(x)@W0a + act(feat)@W0b)
#pragma unroll
  for (int t = 0; t < 4; ++t) acc[t] = (v8f){0.f, 0.f, 0.f, 0.f, 0.f, 0.f, 0.f, 0.f};
  mm_frag<2, 4, false>(xb,   sW + W0A_OFF, lane, acc);  // xb already activated
  mm_frag<3, 4, true >(feat, sW + W0B_OFF, lane, acc);  // feat raw -> act on load
  store_act64(acc, b0, xb, lane);

  // layers 1..3: 64x64
#pragma unroll
  for (int t = 0; t < 4; ++t) acc[t] = (v8f){0.f, 0.f, 0.f, 0.f, 0.f, 0.f, 0.f, 0.f};
  mm_frag<2, 4, false>(xb, sW + W1_OFF, lane, acc);
  store_act64(acc, b1, xb, lane);

#pragma unroll
  for (int t = 0; t < 4; ++t) acc[t] = (v8f){0.f, 0.f, 0.f, 0.f, 0.f, 0.f, 0.f, 0.f};
  mm_frag<2, 4, false>(xb, sW + W2_OFF, lane, acc);
  store_act64(acc, b2, xb, lane);

#pragma unroll
  for (int t = 0; t < 4; ++t) acc[t] = (v8f){0.f, 0.f, 0.f, 0.f, 0.f, 0.f, 0.f, 0.f};
  mm_frag<2, 4, false>(xb, sW + W3_OFF, lane, acc);
  store_act64(acc, b3, xb, lane);

  // out layer: act(x) @ W_out + b_out, only N tile 0 cols 0..2 valid
  v8f co = (v8f){0.f, 0.f, 0.f, 0.f, 0.f, 0.f, 0.f, 0.f};
  mm_frag<2, 1, false>(xb, sW + WOUT_OFF, lane, &co);
  {
    int n0 = lane & 15, g = lane >> 4;
    if (n0 < 3) {
      float bv = b_out[n0];
#pragma unroll
      for (int r = 0; r < 8; ++r) {
        int mr = r + g * 8;
        int orow = row0 + mr;
        if (orow < N) out[orow * 3 + n0] = co[r] + bv;
      }
    }
  }
}

// ---------------- launch ------------------------------------------------------
extern "C" void kernel_launch(void* const* d_in, const int* in_sizes, int n_in,
                              void* d_out, int out_size, void* d_ws, size_t ws_size,
                              hipStream_t stream) {
  const float* p      = (const float*)d_in[0];
  const float* code   = (const float*)d_in[1];
  const float* basis  = (const float*)d_in[2];
  const float* W_init = (const float*)d_in[3];
  const float* b_init = (const float*)d_in[4];
  const float* W0     = (const float*)d_in[5];
  const float* b0     = (const float*)d_in[6];
  const float* W1     = (const float*)d_in[7];
  const float* b1     = (const float*)d_in[8];
  const float* W2     = (const float*)d_in[9];
  const float* b2     = (const float*)d_in[10];
  const float* W3     = (const float*)d_in[11];
  const float* b3     = (const float*)d_in[12];
  const float* W_out  = (const float*)d_in[13];
  const float* b_out  = (const float*)d_in[14];
  float* out = (float*)d_out;
  half_t* ws = (half_t*)d_ws;

  int N = in_sizes[0] / 3;

  prep_weights_kernel<<<(WTOT + 255) / 256, 256, 0, stream>>>(
      W_init, W0, W1, W2, W3, W_out, ws);

  int nblocks = (N + TILE_ROWS - 1) / TILE_ROWS;
  pad_mlp_kernel<<<nblocks, 256, 0, stream>>>(
      p, code, basis, b_init, b0, b1, b2, b3, b_out, ws, out, N);
}
